// InfoVaeLoss_64750926955191
// MI455X (gfx1250) — matrix operand order repaired and checked by reading, not compiled, not run on hardware
//
#include <hip/hip_runtime.h>
#include <hip/hip_bf16.h>

typedef __attribute__((ext_vector_type(16))) __bf16 v16bf;
typedef __attribute__((ext_vector_type(8)))  float  v8f;

#define D_DIM 128

__device__ __forceinline__ unsigned short f32_to_bf16_rne(float f) {
  unsigned int u = __float_as_uint(f);
  u += 0x7FFFu + ((u >> 16) & 1u);   // round-to-nearest-even
  return (unsigned short)(u >> 16);
}

// ---------------------------------------------------------------- zero accums
__global__ void zero_ws_kernel(float* ws) {
  if (threadIdx.x < 8) ws[threadIdx.x] = 0.0f;
}

// ------------------------------------------------- recons: sum((a-b)^2), f4 loads
__global__ void sq_diff_sum_kernel(const float4* __restrict__ a,
                                   const float4* __restrict__ b,
                                   float* __restrict__ acc, int n4) {
  float s = 0.0f;
  int stride = gridDim.x * blockDim.x;
  for (int i = blockIdx.x * blockDim.x + threadIdx.x; i < n4; i += stride) {
    float4 av = a[i], bv = b[i];
    float dx = av.x - bv.x, dy = av.y - bv.y, dz = av.z - bv.z, dw = av.w - bv.w;
    s += dx * dx + dy * dy + dz * dz + dw * dw;
  }
  for (int off = 16; off > 0; off >>= 1) s += __shfl_xor(s, off, 32);
  __shared__ float smem[8];
  int lane = threadIdx.x & 31, wid = threadIdx.x >> 5;
  if (lane == 0) smem[wid] = s;
  __syncthreads();
  if (threadIdx.x == 0) {
    float tot = 0.0f;
    int nw = blockDim.x >> 5;
    for (int w = 0; w < nw; ++w) tot += smem[w];
    atomicAdd(acc, tot);
  }
}

// ---------------------------------------------- kld: sum(1 + lv - mu^2 - exp(lv))
__global__ void kld_sum_kernel(const float* __restrict__ mu,
                               const float* __restrict__ lv,
                               float* __restrict__ acc, int n) {
  float s = 0.0f;
  int stride = gridDim.x * blockDim.x;
  for (int i = blockIdx.x * blockDim.x + threadIdx.x; i < n; i += stride) {
    float m = mu[i], l = lv[i];
    s += 1.0f + l - m * m - __expf(l);
  }
  for (int off = 16; off > 0; off >>= 1) s += __shfl_xor(s, off, 32);
  __shared__ float smem[8];
  int lane = threadIdx.x & 31, wid = threadIdx.x >> 5;
  if (lane == 0) smem[wid] = s;
  __syncthreads();
  if (threadIdx.x == 0) {
    float tot = 0.0f;
    int nw = blockDim.x >> 5;
    for (int w = 0; w < nw; ++w) tot += smem[w];
    atomicAdd(acc, tot);
  }
}

// --------------------- per-row: fp32 -> bf16 convert + fp32 row sum-of-squares
// one block (128 threads) per row of [N, 128]
__global__ void cvt_norm_kernel(const float* __restrict__ src,
                                unsigned short* __restrict__ dst,
                                float* __restrict__ nrm) {
  int row = blockIdx.x;
  int t = threadIdx.x;                       // 0..127 == column
  float v = src[row * D_DIM + t];
  dst[row * D_DIM + t] = f32_to_bf16_rne(v);
  float s = v * v;
  for (int off = 16; off > 0; off >>= 1) s += __shfl_xor(s, off, 32);
  __shared__ float smem[4];
  if ((t & 31) == 0) smem[t >> 5] = s;
  __syncthreads();
  if (t == 0) nrm[row] = smem[0] + smem[1] + smem[2] + smem[3];
}

// ------------------------------------------------------------------ MMD core
// Computes sum over the NxN RBF kernel matrix of exp(-(||a||^2+||b||^2-2 a.b)/(D*sigma))
// fused into a bf16 WMMA GEMM. Block = 128 threads = 4 waves; each wave owns one
// 16x16 C tile (block tile = 16 rows x 64 cols). K = 128 -> 4 x v_wmma_f32_16x16x32_bf16.
union FragBF { uint4 u4[2]; v16bf v; };

__global__ void __launch_bounds__(128)
rbf_mmd_wmma_kernel(const unsigned short* __restrict__ A,   // [N,128] bf16 row-major
                    const unsigned short* __restrict__ B,   // [N,128] bf16 row-major
                    const float* __restrict__ nA,           // [N] fp32 row norms
                    const float* __restrict__ nB,
                    float* __restrict__ acc) {
  const int wave = threadIdx.x >> 5;
  const int lane = threadIdx.x & 31;
  const int h = lane >> 4;                 // half-wave selector
  const int r = lane & 15;                 // row (A) / col (B) within tile
  const int m0 = blockIdx.y * 16;
  const int n0 = blockIdx.x * 64 + wave * 16;

  const unsigned short* arow = A + (m0 + r) * D_DIM;  // A row m0+r
  const unsigned short* brow = B + (n0 + r) * D_DIM;  // B col n0+r (row of B matrix)

  v8f c = {};
#pragma unroll
  for (int kc = 0; kc < D_DIM; kc += 32) {
    FragBF af, bf;
    // A 16x32 bf16 layout: lanes h=0 -> K[kc..kc+7], K[kc+16..kc+23]
    //                      lanes h=1 -> K[kc+8..kc+15], K[kc+24..kc+31]
    af.u4[0] = *(const uint4*)(arow + kc + 8 * h);
    af.u4[1] = *(const uint4*)(arow + kc + 16 + 8 * h);
    // B 32x16 bf16 layout: same K order as A (h=0 -> K[kc..kc+7], K[kc+16..kc+23]; h=1 -> K[kc+8..kc+15], K[kc+24..kc+31])
    bf.u4[0] = *(const uint4*)(brow + kc + 8 * h);
    bf.u4[1] = *(const uint4*)(brow + kc + 16 + 8 * h);
    c = __builtin_amdgcn_wmma_f32_16x16x32_bf16(
        /*neg_a=*/false, af.v, /*neg_b=*/false, bf.v,
        /*c_mod=*/(short)0, c, /*reuse_a=*/false, /*reuse_b=*/false);
  }

  // Fused epilogue: k = exp(sq * -1/(D*sigma)); D*sigma = 128*512 = 65536
  const float negInv = -1.0f / 65536.0f;
  const float nb = nB[n0 + r];             // C col for this lane = n0 + (lane&15)
  float s = 0.0f;
#pragma unroll
  for (int i = 0; i < 8; ++i) {
    float na = nA[m0 + i + 8 * h];         // C row for element i = m0 + i + 8h
    float sq = na + nb - 2.0f * c[i];
    s += __expf(sq * negInv);
  }
  for (int off = 16; off > 0; off >>= 1) s += __shfl_xor(s, off, 32);
  if (lane == 0) atomicAdd(acc, s);
}

// ------------------------------------------------------------------ finalize
__global__ void finalize_kernel(const float* __restrict__ ws, float* __restrict__ out,
                                float inv_img_n, float inv_nn, float n_inv,
                                float bias_inv) {
  if (threadIdx.x != 0 || blockIdx.x != 0) return;
  float recons = ws[0] * inv_img_n;
  float kld = -0.5f * ws[1] * n_inv;                      // mean_N(-0.5 * rowsum)
  float mmd = (ws[2] + ws[3] - 2.0f * ws[4]) * inv_nn;    // k_pp + k_zz - 2 k_pz
  // BETA=5, (1-ALPHA)*kld_weight = 1.5/N, (ALPHA+REG_WEIGHT-1)/bias_corr = 98.5/(N^2-N)
  float loss = 5.0f * recons + 1.5f * n_inv * kld + 98.5f * bias_inv * mmd;
  out[0] = loss;
  out[1] = recons;
  out[2] = mmd;
  out[3] = -kld;
}

extern "C" void kernel_launch(void* const* d_in, const int* in_sizes, int n_in,
                              void* d_out, int out_size, void* d_ws, size_t ws_size,
                              hipStream_t stream) {
  const float* recons  = (const float*)d_in[0];
  const float* x       = (const float*)d_in[1];
  const float* z       = (const float*)d_in[2];
  const float* mu      = (const float*)d_in[3];
  const float* log_var = (const float*)d_in[4];
  const float* prior_z = (const float*)d_in[5];

  const int img_n = in_sizes[0];           // 4096*3*64*64
  const int zn    = in_sizes[2];           // N * 128
  const int N     = zn / D_DIM;            // 4096

  // workspace layout (floats): [0..7] accumulators | nrm_p[N] | nrm_z[N] | bf16 p | bf16 z
  float* ws    = (float*)d_ws;
  float* acc   = ws;
  float* nrm_p = ws + 16;
  float* nrm_z = ws + 16 + N;
  unsigned short* p_bf = (unsigned short*)(ws + 16 + 2 * N);
  unsigned short* z_bf = p_bf + (size_t)N * D_DIM;

  size_t need = (16 + 2 * (size_t)N) * sizeof(float) + 2 * (size_t)N * D_DIM * sizeof(unsigned short);
  if (ws_size < need) return;

  zero_ws_kernel<<<1, 32, 0, stream>>>(acc);

  // HBM-bound term: 402 MB of fp32 reads via b128 loads
  sq_diff_sum_kernel<<<2048, 256, 0, stream>>>((const float4*)recons, (const float4*)x,
                                               &acc[0], img_n / 4);
  kld_sum_kernel<<<512, 256, 0, stream>>>(mu, log_var, &acc[1], zn);

  // bf16 conversion + fp32 row norms (inputs stay untouched; scratch in ws)
  cvt_norm_kernel<<<N, D_DIM, 0, stream>>>(prior_z, p_bf, nrm_p);
  cvt_norm_kernel<<<N, D_DIM, 0, stream>>>(z,       z_bf, nrm_z);

  // three fused WMMA RBF-kernel-sum GEMMs: (p,p), (z,z), (p,z)
  dim3 grid(N / 64, N / 16);
  rbf_mmd_wmma_kernel<<<grid, 128, 0, stream>>>(p_bf, p_bf, nrm_p, nrm_p, &acc[2]);
  rbf_mmd_wmma_kernel<<<grid, 128, 0, stream>>>(z_bf, z_bf, nrm_z, nrm_z, &acc[3]);
  rbf_mmd_wmma_kernel<<<grid, 128, 0, stream>>>(p_bf, z_bf, nrm_p, nrm_z, &acc[4]);

  finalize_kernel<<<1, 1, 0, stream>>>(acc, (float*)d_out,
                                       1.0f / (float)img_n,
                                       1.0f / ((float)N * (float)N),
                                       1.0f / (float)N,
                                       1.0f / ((float)N * (float)N - (float)N));
}
